// modeler_10960756539513
// MI455X (gfx1250) — matrix operand order repaired and checked by reading, not compiled, not run on hardware
//
#include <hip/hip_runtime.h>
#include <math.h>

// ---------------------------------------------------------------------------
// HAN-style 2-layer hetero-GNN for MI455X (gfx1250, wave32).
//   SpMM: edge-parallel wave32 + global_atomic_add_f32 (L2-resident gathers)
//   GEMM: v_wmma_f32_16x16x32_bf16, fully unrolled K (template-specialized)
//   SemAtt: wave-per-node, lane==attention-column (ATT_H==32), shfl reduce
// ---------------------------------------------------------------------------

typedef __bf16 bf16_t;
typedef __attribute__((ext_vector_type(16))) __bf16 v16bf;
typedef __attribute__((ext_vector_type(8)))  __bf16 v8bf;
typedef __attribute__((ext_vector_type(8)))  float  v8f;

constexpr int kNP  = 200000;
constexpr int kNA  = 100000;
constexpr int kNS  = 50000;
constexpr int kEPA = 1600000;
constexpr int kEPS = 400000;
constexpr int kEAP = 1600000;
constexpr int kESP = 400000;
constexpr int kD   = 128;      // EMB == HID == HID2 == OUT

// ---------------------------------------------------------------------------
// zero-fill (graph-capture-safe memset replacement)
// ---------------------------------------------------------------------------
__global__ void han_zero(float* __restrict__ p, long n) {
  long i = blockIdx.x * (long)blockDim.x + threadIdx.x;
  if (i < n) p[i] = 0.0f;
}

// ---------------------------------------------------------------------------
// weight prep: f32 row-major [K][128] -> bf16 col-major [128][K]
// (makes each 8-elem half of the WMMA B-fragment a contiguous 16B v8bf load per lane)
// ---------------------------------------------------------------------------
__global__ void han_prep_w(const float* __restrict__ W, bf16_t* __restrict__ Wt, int K) {
  int i = blockIdx.x * blockDim.x + threadIdx.x;
  if (i >= K * kD) return;
  int k = i / kD, n = i % kD;
  Wt[(long)n * K + k] = (bf16_t)W[(long)k * kD + n];
}

// ---------------------------------------------------------------------------
// SpMM: out[r] += w[e] * X[c[e]]   (one wave32 per edge, 4 floats per lane)
// ---------------------------------------------------------------------------
__global__ void han_spmm(const int* __restrict__ rows, const int* __restrict__ cols,
                         const float* __restrict__ w, const void* __restrict__ X, int xf32,
                         float* __restrict__ Out, int nE) {
  int gw   = (int)((blockIdx.x * (long)blockDim.x + threadIdx.x) >> 5);
  int lane = threadIdx.x & 31;
  if (gw >= nE) return;
  int   r  = rows[gw];
  int   c  = cols[gw];
  float wt = w[gw];
  long si = (long)c * kD + lane * 4;
  float v[4];
  if (xf32) {
    const float* x = (const float*)X + si;
#pragma unroll
    for (int i = 0; i < 4; ++i) v[i] = x[i];
  } else {
    const bf16_t* x = (const bf16_t*)X + si;
#pragma unroll
    for (int i = 0; i < 4; ++i) v[i] = (float)x[i];
  }
  float* o = Out + (long)r * kD + lane * 4;
#pragma unroll
  for (int i = 0; i < 4; ++i) atomicAdd(o + i, wt * v[i]);
}

// ---------------------------------------------------------------------------
// WMMA GEMM:  Y = act( [X1|X2] @ W (+ bias) ),  W given col-major bf16.
// All shape/type decisions are template parameters so the K-loop fully
// unrolls into back-to-back v_wmma with no branches in the hot path.
//
// Block = 256 threads = 8 waves; block owns 16 rows, wave w owns cols [16w,16w+16).
// A fragment (16x32 bf16): lane M = lane%16; elems 0..7 -> K = k0+half*8+i,
//                          elems 8..15 -> K = k0+16+half*8+i  (half = lane/16).
// B fragment (32x16 bf16): lane N = lane%16; same K mapping as A fragment.
// C/D (16x16 f32): elem j -> row M = j + 8*half, col N = lane%16.
// ---------------------------------------------------------------------------
template <bool F32SRC>
__device__ __forceinline__ v8bf han_load8(const void* X, long base) {
  v8bf r;
  if (F32SRC) {
    const float* f = (const float*)X + base;
#pragma unroll
    for (int i = 0; i < 8; ++i) r[i] = (bf16_t)f[i];
  } else {
    r = *(const v8bf*)((const bf16_t*)X + base);
  }
  return r;
}

template <int K1, bool F1, int K2, bool F2, int ACT, bool YF32, bool BIAS>
__global__ void han_gemm_t(const void* __restrict__ X1, const void* __restrict__ X2,
                           const bf16_t* __restrict__ Wt,   // col-major [128][K1+K2]
                           const float* __restrict__ bias,
                           void* __restrict__ Y) {
  constexpr int K   = K1 + K2;
  const int lane    = threadIdx.x & 31;
  const int wave    = threadIdx.x >> 5;
  const int rowBase = blockIdx.x * 16;
  const int colBase = wave * 16;
  const int m       = lane & 15;
  const int half    = lane >> 4;
  const int row     = rowBase + m;
  const int n       = colBase + m;

  v8f acc = {};
#pragma unroll
  for (int k0 = 0; k0 < K; k0 += 32) {
    // Both 8-elem chunks of a 32-wide K-step always come from the same source
    // (K1, K2 are multiples of 32), and k0 is a compile-time constant.
    v8bf a0, a1;
    if (k0 < K1) {
      a0 = han_load8<F1>(X1, (long)row * K1 + (k0 + half * 8));
      a1 = han_load8<F1>(X1, (long)row * K1 + (k0 + 16 + half * 8));
    } else {
      a0 = han_load8<F2>(X2, (long)row * K2 + (k0 - K1 + half * 8));
      a1 = han_load8<F2>(X2, (long)row * K2 + (k0 - K1 + 16 + half * 8));
    }
    v16bf a;
#pragma unroll
    for (int i = 0; i < 8; ++i) { a[i] = a0[i]; a[i + 8] = a1[i]; }
    v8bf b0 = *(const v8bf*)(Wt + (long)n * K + k0 + half * 8);
    v8bf b1 = *(const v8bf*)(Wt + (long)n * K + k0 + 16 + half * 8);
    v16bf b;
#pragma unroll
    for (int i = 0; i < 8; ++i) { b[i] = b0[i]; b[i + 8] = b1[i]; }
    acc = __builtin_amdgcn_wmma_f32_16x16x32_bf16(
        false, a, false, b, (short)0, acc, false, false);
  }

  float bv = BIAS ? bias[n] : 0.0f;
#pragma unroll
  for (int j = 0; j < 8; ++j) {
    float v = acc[j] + bv;
    if (ACT == 1)      v = v > 0.0f ? v : 0.0f;        // relu
    else if (ACT == 2) v = v > 0.0f ? v : expm1f(v);   // elu
    long oi = (long)(rowBase + j + 8 * half) * kD + n;
    if (YF32) ((float*)Y)[oi]  = v;
    else      ((bf16_t*)Y)[oi] = (bf16_t)v;
  }
}

// ---------------------------------------------------------------------------
// Semantic attention, stage 1: per-node h = q . tanh(v @ AW + b), summed into
// a per-relation scalar.  One wave per node; lane j == attention column j.
// ---------------------------------------------------------------------------
__global__ void han_att_partial(const bf16_t* __restrict__ V, const float* __restrict__ AW,
                                const float* __restrict__ Ab, const float* __restrict__ Aq,
                                float* __restrict__ accum, int nNodes) {
  int gw   = (int)((blockIdx.x * (long)blockDim.x + threadIdx.x) >> 5);
  int lane = threadIdx.x & 31;
  if (gw >= nNodes) return;
  float s = Ab[lane];
  const bf16_t* v = V + (long)gw * kD;
  for (int k = 0; k < kD; ++k) s += (float)v[k] * AW[k * 32 + lane];
  float h = tanhf(s) * Aq[lane];
#pragma unroll
  for (int off = 16; off > 0; off >>= 1) h += __shfl_xor(h, off, 32);
  if (lane == 0) atomicAdd(accum, h);
}

// stage 2: beta = softmax(mean_h over nodes), 2 relations
__global__ void han_beta(const float* __restrict__ partial, float* __restrict__ beta,
                         float invN) {
  float m0 = partial[0] * invN, m1 = partial[1] * invN;
  float mx = fmaxf(m0, m1);
  float e0 = expf(m0 - mx), e1 = expf(m1 - mx);
  float s  = e0 + e1;
  beta[0] = e0 / s;
  beta[1] = e1 / s;
}

// stage 3: Out = beta0*V0 + beta1*V1   (elementwise, in-place safe)
__global__ void han_combine(const bf16_t* __restrict__ V0, const bf16_t* __restrict__ V1,
                            const float* __restrict__ beta, bf16_t* __restrict__ Out, long n) {
  long i = blockIdx.x * (long)blockDim.x + threadIdx.x;
  if (i >= n) return;
  Out[i] = (bf16_t)(beta[0] * (float)V0[i] + beta[1] * (float)V1[i]);
}

// ---------------------------------------------------------------------------
// Orchestration
// ---------------------------------------------------------------------------
extern "C" void kernel_launch(void* const* d_in, const int* in_sizes, int n_in,
                              void* d_out, int out_size, void* d_ws, size_t ws_size,
                              hipStream_t stream) {
  (void)in_sizes; (void)n_in; (void)out_size; (void)ws_size;

  // --- inputs (setup_inputs() dict order) ---
  const float* ft_p = (const float*)d_in[0];
  const float* ft_a = (const float*)d_in[1];
  const float* ft_s = (const float*)d_in[2];
  const int* r_pa = (const int*)d_in[3];  const int* c_pa = (const int*)d_in[4];
  const int* r_ps = (const int*)d_in[5];  const int* c_ps = (const int*)d_in[6];
  const int* r_ap = (const int*)d_in[7];  const int* c_ap = (const int*)d_in[8];
  const int* r_sp = (const int*)d_in[9];  const int* c_sp = (const int*)d_in[10];
  const float* w_pa = (const float*)d_in[11];
  const float* w_ps = (const float*)d_in[12];
  const float* w_ap = (const float*)d_in[13];
  const float* w_sp = (const float*)d_in[14];
  const float* W0_pa = (const float*)d_in[15]; const float* W1_pa = (const float*)d_in[16];
  const float* W0_ps = (const float*)d_in[17]; const float* W1_ps = (const float*)d_in[18];
  const float* W0_ap = (const float*)d_in[19]; const float* W1_ap = (const float*)d_in[20];
  const float* W0_sp = (const float*)d_in[21]; const float* W1_sp = (const float*)d_in[22];
  const float* A0_W = (const float*)d_in[23]; const float* A0_b = (const float*)d_in[24];
  const float* A0_q = (const float*)d_in[25];
  const float* A1_W = (const float*)d_in[26]; const float* A1_b = (const float*)d_in[27];
  const float* A1_q = (const float*)d_in[28];
  const float* Wfc_p = (const float*)d_in[29]; const float* bfc_p = (const float*)d_in[30];
  const float* Wfc_a = (const float*)d_in[31]; const float* bfc_a = (const float*)d_in[32];
  const float* Wfc_s = (const float*)d_in[33]; const float* bfc_s = (const float*)d_in[34];

  // --- workspace carving (~295 MB) ---
  size_t off = 0;
  auto take = [&](size_t bytes) -> char* {
    off = (off + 255) & ~(size_t)255;
    char* p = (char*)d_ws + off;
    off += bytes;
    return p;
  };
  float*  agg = (float*)take((size_t)kNP * kD * 4);   // f32 atomic SpMM scratch
  bf16_t* v0  = (bf16_t*)take((size_t)kNP * kD * 2);  // v_pa / u_pa / e2_p
  bf16_t* v1  = (bf16_t*)take((size_t)kNP * kD * 2);  // v_ps / u_ps
  bf16_t* e1p = (bf16_t*)take((size_t)kNP * kD * 2);  // e1_p
  bf16_t* e1a = (bf16_t*)take((size_t)kNA * kD * 2);  // e1_a -> e2_a
  bf16_t* e1s = (bf16_t*)take((size_t)kNS * kD * 2);  // e1_s -> e2_s
  bf16_t* wt0  = (bf16_t*)take(16384 * 2);
  bf16_t* wt1  = (bf16_t*)take(16384 * 2);
  bf16_t* wt2  = (bf16_t*)take(16384 * 2);
  bf16_t* wt3  = (bf16_t*)take(16384 * 2);
  bf16_t* wt4  = (bf16_t*)take(16384 * 2);
  bf16_t* wt5  = (bf16_t*)take(16384 * 2);
  bf16_t* wt6  = (bf16_t*)take(16384 * 2);
  bf16_t* wt7  = (bf16_t*)take(16384 * 2);
  bf16_t* wt8  = (bf16_t*)take(32768 * 2);
  bf16_t* wt9  = (bf16_t*)take(32768 * 2);
  bf16_t* wt10 = (bf16_t*)take(32768 * 2);
  float*  part = (float*)take(16 * 4);  // [0..3] attn sums, [4..5],[6..7] betas

  // --- launch helpers ---
  auto zero = [&](float* p, long n) {
    han_zero<<<(unsigned)((n + 255) / 256), 256, 0, stream>>>(p, n);
  };
  auto prep = [&](const float* W, bf16_t* Wt, int K) {
    han_prep_w<<<(K * kD + 255) / 256, 256, 0, stream>>>(W, Wt, K);
  };
  auto spmm = [&](const int* r, const int* c, const float* w, const void* X, int xf32,
                  float* out, int nE) {
    han_spmm<<<(nE + 7) / 8, 256, 0, stream>>>(r, c, w, X, xf32, out, nE);
  };
  // f32 (N,128) @ W -> relu -> bf16 (N,128)
  auto gemm128 = [&](const float* X, const bf16_t* Wt, bf16_t* Y, int N) {
    han_gemm_t<128, true, 0, false, 1, false, false>
        <<<N / 16, 256, 0, stream>>>(X, nullptr, Wt, nullptr, Y);
  };
  // concat(bf16 (N,128), f32 (N,128)) @ Wfc + b -> elu -> f32 (N,128)
  auto gemmfc = [&](const bf16_t* Xb, const float* Xf, const bf16_t* Wt,
                    const float* bias, float* Y, int N) {
    han_gemm_t<128, false, 128, true, 2, true, true>
        <<<N / 16, 256, 0, stream>>>(Xb, Xf, Wt, bias, Y);
  };
  auto attp = [&](const bf16_t* V, const float* AW, const float* Ab, const float* Aq,
                  float* acc, int N) {
    han_att_partial<<<(N + 7) / 8, 256, 0, stream>>>(V, AW, Ab, Aq, acc, N);
  };
  auto comb = [&](const bf16_t* A, const bf16_t* B, const float* bet, bf16_t* Out, long n) {
    han_combine<<<(unsigned)((n + 255) / 256), 256, 0, stream>>>(A, B, bet, Out, n);
  };

  // --- weight prep (once per call; cheap) ---
  prep(W0_pa, wt0, 128); prep(W0_ps, wt1, 128); prep(W0_ap, wt2, 128); prep(W0_sp, wt3, 128);
  prep(W1_pa, wt4, 128); prep(W1_ps, wt5, 128); prep(W1_ap, wt6, 128); prep(W1_sp, wt7, 128);
  prep(Wfc_p, wt8, 256); prep(Wfc_a, wt9, 256); prep(Wfc_s, wt10, 256);
  zero(part, 8);

  // --- layer 0, type P ---
  zero(agg, (long)kNP * kD);
  spmm(r_pa, c_pa, w_pa, ft_a, 1, agg, kEPA);
  gemm128(agg, wt0, v0, kNP);                                             // v_pa
  zero(agg, (long)kNP * kD);
  spmm(r_ps, c_ps, w_ps, ft_s, 1, agg, kEPS);
  gemm128(agg, wt1, v1, kNP);                                             // v_ps
  attp(v0, A0_W, A0_b, A0_q, part + 0, kNP);
  attp(v1, A0_W, A0_b, A0_q, part + 1, kNP);
  han_beta<<<1, 1, 0, stream>>>(part, part + 4, 1.0f / (float)kNP);
  comb(v0, v1, part + 4, e1p, (long)kNP * kD);                            // e1_p

  // --- layer 0, types A and S ---
  zero(agg, (long)kNA * kD);
  spmm(r_ap, c_ap, w_ap, ft_p, 1, agg, kEAP);
  gemm128(agg, wt2, e1a, kNA);                                            // e1_a
  zero(agg, (long)kNS * kD);
  spmm(r_sp, c_sp, w_sp, ft_p, 1, agg, kESP);
  gemm128(agg, wt3, e1s, kNS);                                            // e1_s

  // --- layer 1, type P ---
  zero(agg, (long)kNP * kD);
  spmm(r_pa, c_pa, w_pa, e1a, 0, agg, kEPA);
  gemm128(agg, wt4, v0, kNP);                                             // u_pa
  zero(agg, (long)kNP * kD);
  spmm(r_ps, c_ps, w_ps, e1s, 0, agg, kEPS);
  gemm128(agg, wt5, v1, kNP);                                             // u_ps
  attp(v0, A1_W, A1_b, A1_q, part + 2, kNP);
  attp(v1, A1_W, A1_b, A1_q, part + 3, kNP);
  han_beta<<<1, 1, 0, stream>>>(part + 2, part + 6, 1.0f / (float)kNP);
  comb(v0, v1, part + 6, v0, (long)kNP * kD);                             // e2_p (in-place)

  // --- layer 1, types A and S (use e1_p) ---
  zero(agg, (long)kNA * kD);
  spmm(r_ap, c_ap, w_ap, e1p, 0, agg, kEAP);
  gemm128(agg, wt6, e1a, kNA);                                            // e2_a
  zero(agg, (long)kNS * kD);
  spmm(r_sp, c_sp, w_sp, e1p, 0, agg, kESP);
  gemm128(agg, wt7, e1s, kNS);                                            // e2_s

  // --- FC: elu(concat([e2, ft]) @ Wfc + bfc) -> d_out (f32) ---
  float* out = (float*)d_out;
  gemmfc(v0,  ft_p, wt8,  bfc_p, out,                          kNP);
  gemmfc(e1a, ft_a, wt9,  bfc_a, out + (long)kNP * kD,         kNA);
  gemmfc(e1s, ft_s, wt10, bfc_s, out + (long)(kNP + kNA) * kD, kNS);
}
